// Argphy_12455405158451
// MI455X (gfx1250) — matrix operand order repaired and checked by reading, not compiled, not run on hardware
//
#include <hip/hip_runtime.h>

typedef _Float16 half_t;
typedef __attribute__((ext_vector_type(16))) _Float16 v16h;
typedef __attribute__((ext_vector_type(8)))  float    v8f;

#define NWG      32
#define TPB      128
#define HIDDEN   256
#define WIDTH    1024
#define IN0      273      // HIDDEN + 17
#define IN0P     288      // padded to multiple of 32
#define KC0      9        // IN0P / 32
#define KCW      32       // WIDTH / 32
#define TSTEPS   64
#define SUBSTEPS 8

// LDS layout (in half_t units): pre-swizzled WMMA A-fragments of the weight slices
#define W0_FRAG  (2*KC0*512)              //  9216 halves : 32 rows of W0 (K padded to 288)
#define WH_FRAG  (2*KCW*512)              // 32768 halves : 32 rows of one Wh layer
#define WL_FRAG  (KCW*512)                // 16384 halves : 16 rows of Wl (only wg<16)
#define W_TOTAL  (W0_FRAG + 3*WH_FRAG + WL_FRAG)   // 123904 halves
#define XFRAG_H  (KCW*512)                // B-fragment image of x (col 0 live, rest zero)
#define SMEM_HALF_BYTES ((W_TOTAL + XFRAG_H)*2)    // 280576 B
#define SMEM_FLOATS (IN0P + IN0 + 4*IN0 + 64 + TPB + 8)
#define SMEM_BYTES  (SMEM_HALF_BYTES + SMEM_FLOATS*4)

// CDNA5 A-fragment (16-bit, 16x32) K position for (lane, half-slot i):
// lanes 0-15: V0..V3 -> K 0..7, V4..V7 -> K 16..23 ; lanes 16-31: +8
__device__ __forceinline__ int kmapA(int L, int i) {
  int v = i >> 1, p = i & 1, hi = L >> 4;
  return ((v < 4) ? 0 : 16) + hi * 8 + ((v & 3) << 1) + p;
}

// B-fragment (16-bit, 32x16, mirrors the documented 16-bit C/D packing):
// column N=0 lives in lane 0 (K 0..7, 16..23) and lane 16 (K 8..15, 24..31).
__device__ __forceinline__ void xput(half_t* xfrag, int k, float v) {
  int ck = k >> 5, j = k & 31;
  xfrag[ck * 512 + (((j >> 3) & 1) << 8) + ((j >> 4) << 3) + (j & 7)] = (half_t)v;
}

__device__ __forceinline__ float softplus_f(float x) {
  return (x > 20.f) ? x : log1pf(expf(x));
}

// One layer slice: NT row-tiles of 16, K split between wave pairs.
// Compile-time trip counts -> fully unrolled straight-line load/WMMA stream so
// the backend can prefetch LDS chunks ahead (partial s_wait_dscnt) and fill
// WMMA hazard slots with real work instead of v_nop.
template <int KC, int NT>
__device__ __forceinline__ void mm_tiles(const half_t* __restrict__ wbase,
                                         const half_t* __restrict__ xfrag,
                                         float* pacc, int wave, int lane,
                                         int tile, int khalf) {
  if (tile < NT) {
    constexpr int H0 = (KC + 1) / 2;
    v8f acc = {0.f, 0.f, 0.f, 0.f, 0.f, 0.f, 0.f, 0.f};
    const half_t* wt = wbase + tile * (KC * 512) + lane * 16;
    const half_t* xb = xfrag + lane * 16;
    if (khalf == 0) {
#pragma unroll
      for (int ck = 0; ck < H0; ++ck) {
        v16h a = *(const v16h*)(wt + ck * 512);
        v16h b = *(const v16h*)(xb + ck * 512);
        acc = __builtin_amdgcn_wmma_f32_16x16x32_f16(false, a, false, b,
                                                     (short)0, acc, false, false);
      }
    } else {
#pragma unroll
      for (int ck = H0; ck < KC; ++ck) {
        v16h a = *(const v16h*)(wt + ck * 512);
        v16h b = *(const v16h*)(xb + ck * 512);
        acc = __builtin_amdgcn_wmma_f32_16x16x32_f16(false, a, false, b,
                                                     (short)0, acc, false, false);
      }
    }
    if ((lane & 15) == 0) {
      int hs = lane >> 4;
#pragma unroll
      for (int r = 0; r < 8; ++r) pacc[wave * 16 + hs * 8 + r] = acc[r];
    }
  }
}

__global__ void init_ws(unsigned* cnt) { if (threadIdx.x == 0) *cnt = 0u; }

__global__ __launch_bounds__(TPB, 1)
void neuralode_persistent(const float* __restrict__ ts,  const float* __restrict__ W0,
                          const float* __restrict__ b0,  const float* __restrict__ Wh,
                          const float* __restrict__ bh,  const float* __restrict__ Wl,
                          const float* __restrict__ bl,  const float* __restrict__ betaW,
                          const float* __restrict__ betaB, const float* __restrict__ hvec,
                          const float* __restrict__ scale, const float* __restrict__ y0log,
                          float* __restrict__ out, void* __restrict__ ws) {
  extern __shared__ char smem[];
  half_t* ldsW  = (half_t*)smem;
  half_t* xfrag = ldsW + W_TOTAL;
  float*  xe    = (float*)(smem + SMEM_HALF_BYTES);
  float*  yloc  = xe + IN0P;          // local replica of full ODE state (273)
  float*  kloc  = yloc + IN0;         // k1..k4 (4 x 273)
  float*  pacc  = kloc + 4 * IN0;     // 64 partial-accumulator slots
  float*  red   = pacc + 64;          // TPB reduction scratch
  float*  shv   = red + TPB;          // shv[1] = scale[0]

  const int tid   = threadIdx.x;
  const int wg    = blockIdx.x;
  const int lane  = tid & 31;
  const int wave  = tid >> 5;
  const int tile  = wave & 1;
  const int khalf = wave >> 1;

  unsigned* cnt = (unsigned*)ws;
  float* zA    = (float*)((char*)ws + 256);
  float* zB    = zA + WIDTH;
  float* kglob = zB + WIDTH;          // 4 * 256 (MLP part of each k)
  unsigned barEp = 0;

#define GBAR() do {                                                          \
    barEp++;                                                                 \
    __syncthreads();                                                         \
    if (tid == 0) {                                                          \
      __threadfence();                                                       \
      atomicAdd(cnt, 1u);                                                    \
      while (*((volatile unsigned*)cnt) < barEp * NWG)                       \
        __builtin_amdgcn_s_sleep(2);                                         \
      __threadfence();                                                       \
    }                                                                        \
    __syncthreads();                                                         \
  } while (0)

  // ---- one-time: quantize f32 weights -> f16, pre-swizzled into A-fragment order
  for (int idx = tid; idx < W0_FRAG; idx += TPB) {
    int i = idx & 15, L = (idx >> 4) & 31;
    int ck = (idx >> 9) % KC0;
    int t  = idx / (KC0 * 512);
    int m  = wg * 32 + t * 16 + (L & 15);
    int k  = ck * 32 + kmapA(L, i);
    ldsW[idx] = (half_t)((k < IN0) ? W0[m * IN0 + k] : 0.0f);
  }
  for (int l = 0; l < 3; ++l) {
    half_t* dst = ldsW + W0_FRAG + l * WH_FRAG;
    const float* src = Wh + (size_t)l * WIDTH * WIDTH;
    for (int idx = tid; idx < WH_FRAG; idx += TPB) {
      int i = idx & 15, L = (idx >> 4) & 31;
      int ck = (idx >> 9) & 31, t = idx >> 14;
      int m  = wg * 32 + t * 16 + (L & 15);
      int k  = ck * 32 + kmapA(L, i);
      dst[idx] = (half_t)src[m * WIDTH + k];
    }
  }
  if (wg < 16) {
    half_t* dst = ldsW + W0_FRAG + 3 * WH_FRAG;
    for (int idx = tid; idx < WL_FRAG; idx += TPB) {
      int i = idx & 15, L = (idx >> 4) & 31;
      int ck = (idx >> 9) & 31;
      int m  = wg * 16 + (L & 15);
      int k  = ck * 32 + kmapA(L, i);
      dst[idx] = (half_t)Wl[m * WIDTH + k];
    }
  }
  for (int idx = tid; idx < XFRAG_H; idx += TPB) xfrag[idx] = (half_t)0.0f;
  for (int i = IN0 + tid; i < IN0P; i += TPB) xe[i] = 0.0f;

  // ---- y0 = [softmax(y0_log); hvec]
  if (tid == 0) {
    float mx = -1e30f;
    for (int j = 0; j < 17; ++j) mx = fmaxf(mx, y0log[j]);
    float s = 0.f, e[17];
    for (int j = 0; j < 17; ++j) { e[j] = expf(y0log[j] - mx); s += e[j]; }
    for (int j = 0; j < 17; ++j) yloc[j] = e[j] / s;
    shv[1] = scale[0];
  }
  for (int j = tid; j < HIDDEN; j += TPB) yloc[17 + j] = hvec[j];
  __syncthreads();

  if (wg == 0) {  // ys[0]
    for (int j = tid; j < 17; j += TPB) out[j] = yloc[j];
    for (int j = tid; j < HIDDEN; j += TPB) out[TSTEPS * 17 + j] = yloc[17 + j];
  }

  const float xi = 13.f/12.f, mu = 0.041f/12.f, sg = 91.f/12.f,
              nu = 36.f/12.f, ga = 1.8f/12.f;

  for (int iv = 0; iv < TSTEPS - 1; ++iv) {
    float dt = (ts[iv + 1] - ts[iv]) / (float)SUBSTEPS;
    for (int ss = 0; ss < SUBSTEPS; ++ss) {
      for (int s = 0; s < 4; ++s) {
        // evaluation point for this RK4 stage
        float cf = (s == 3) ? dt : 0.5f * dt;
        for (int i = tid; i < IN0; i += TPB)
          xe[i] = yloc[i] + ((s == 0) ? 0.f : cf * kloc[(s - 1) * IN0 + i]);
        __syncthreads();

        // build layer-0 B-fragment:  x = [h ; state]
        for (int idx = tid; idx < IN0P; idx += TPB) {
          float v = 0.f;
          if (idx < IN0) v = (idx < HIDDEN) ? xe[17 + idx] : xe[idx - HIDDEN];
          xput(xfrag, idx, v);
        }
        // beta dot (redundant per WG, cheap)
        float ps = 0.f;
        for (int j = tid; j < HIDDEN; j += TPB) ps += betaW[j] * xe[17 + j];
        red[tid] = ps;
        __syncthreads();
        if (tid == 0) {
          float d = 0.f;
          for (int j = 0; j < TPB; ++j) d += red[j];
          float bb1 = 8.f / (1.f + expf(-(d + betaB[0]))) + 25.f;
          float bb2 = .5f*bb1, bb3 = .35f*bb1, bb4 = .25f*bb1;
          float M=xe[0],S1=xe[1],E1=xe[2],E2=xe[3],E3=xe[4],E4=xe[5];
          float I1=xe[6],I2=xe[7],I3=xe[8],I4=xe[9];
          float R1=xe[10],R2=xe[11],R3=xe[12],R4=xe[13];
          float S2=xe[14],S3=xe[15],S4=xe[16];
          float I = I1+I2+I3+I4, R = R1+R2+R3+R4;
          float* d0 = kloc + s * IN0;
          d0[0]=R*mu-(xi+mu)*M;
          d0[1]=mu*(1.f-R)+xi*M-mu*S1-bb1*I*S1;
          d0[2]=bb1*I*S1-(mu+sg)*E1;  d0[3]=bb2*I*S2-(mu+sg)*E2;
          d0[4]=bb3*I*S3-(mu+sg)*E3;  d0[5]=bb4*I*S4-(mu+sg)*E4;
          d0[6]=sg*E1-(nu+mu)*I1;     d0[7]=sg*E2-(nu+mu)*I2;
          d0[8]=sg*E3-(nu+mu)*I3;     d0[9]=sg*E4-(nu+mu)*I4;
          d0[10]=nu*I1-(mu+ga)*R1;    d0[11]=nu*I2-(mu+ga)*R2;
          d0[12]=nu*I3-(mu+ga)*R3;    d0[13]=nu*I4-(mu+ga)*R4;
          d0[14]=ga*R1-mu*S2-bb2*I*S2;
          d0[15]=ga*R2-mu*S3-bb3*I*S3;
          d0[16]=ga*(R3+R4)-mu*S4-bb4*I*S4;
        }
        __syncthreads();

        // layer 0 : 1024x273 (padded 288)
        mm_tiles<KC0, 2>(ldsW, xfrag, pacc, wave, lane, tile, khalf);
        __syncthreads();
        if (tid < 32) {
          int t = tid >> 4, rl = tid & 15;
          float v = pacc[t*16 + rl] + pacc[(t+2)*16 + rl] + b0[wg*32 + tid];
          zA[wg*32 + tid] = softplus_f(v);
        }
        GBAR();

        // layers 1..3 : 1024x1024
        const float* zin = zA; float* zout = zB;
        for (int l = 0; l < 3; ++l) {
          for (int idx = tid; idx < WIDTH; idx += TPB) xput(xfrag, idx, zin[idx]);
          __syncthreads();
          mm_tiles<KCW, 2>(ldsW + W0_FRAG + l*WH_FRAG, xfrag, pacc,
                           wave, lane, tile, khalf);
          __syncthreads();
          if (tid < 32) {
            int t = tid >> 4, rl = tid & 15;
            float v = pacc[t*16 + rl] + pacc[(t+2)*16 + rl] + bh[l*WIDTH + wg*32 + tid];
            zout[wg*32 + tid] = softplus_f(v);
          }
          GBAR();
          const float* tmp = zin; zin = zout; zout = (float*)tmp;
        }
        // final activations are in zin (== zB)

        // last layer : 256x1024, rows owned by WGs 0..15 (16 each)
        if (wg < 16) {
          for (int idx = tid; idx < WIDTH; idx += TPB) xput(xfrag, idx, zin[idx]);
          __syncthreads();
          mm_tiles<KCW, 1>(ldsW + W0_FRAG + 3*WH_FRAG, xfrag, pacc,
                           wave, lane, tile, khalf);
          __syncthreads();
          if (tid < 16) {
            float v = pacc[tid] + pacc[32 + tid] + bl[wg*16 + tid];
            kglob[s*HIDDEN + wg*16 + tid] = shv[1] * tanhf(0.01f * v);
          }
        }
        GBAR();
        for (int j = tid; j < HIDDEN; j += TPB)
          kloc[s*IN0 + 17 + j] = kglob[s*HIDDEN + j];
        __syncthreads();
      } // stages

      // y += dt/6 * (k1 + 2k2 + 2k3 + k4)   (redundant in every WG)
      float c6 = dt / 6.f;
      for (int i = tid; i < IN0; i += TPB)
        yloc[i] += c6 * (kloc[i] + 2.f*kloc[IN0+i] + 2.f*kloc[2*IN0+i] + kloc[3*IN0+i]);
      __syncthreads();
    } // substeps

    if (wg == 0) {
      for (int j = tid; j < 17; j += TPB) out[(iv+1)*17 + j] = yloc[j];
      for (int j = tid; j < HIDDEN; j += TPB)
        out[TSTEPS*17 + (iv+1)*HIDDEN + j] = yloc[17 + j];
    }
  } // intervals
#undef GBAR
}

extern "C" void kernel_launch(void* const* d_in, const int* in_sizes, int n_in,
                              void* d_out, int out_size, void* d_ws, size_t ws_size,
                              hipStream_t stream) {
  (void)in_sizes; (void)n_in; (void)out_size; (void)ws_size;
  const float* ts    = (const float*)d_in[0];
  const float* W0    = (const float*)d_in[1];
  const float* b0    = (const float*)d_in[2];
  const float* Wh    = (const float*)d_in[3];
  const float* bh    = (const float*)d_in[4];
  const float* Wl    = (const float*)d_in[5];
  const float* bl    = (const float*)d_in[6];
  const float* betaW = (const float*)d_in[7];
  const float* betaB = (const float*)d_in[8];
  const float* hvec  = (const float*)d_in[9];
  const float* scale = (const float*)d_in[10];
  const float* y0log = (const float*)d_in[11];

  (void)hipFuncSetAttribute(reinterpret_cast<const void*>(neuralode_persistent),
                            hipFuncAttributeMaxDynamicSharedMemorySize, SMEM_BYTES);
  init_ws<<<1, 64, 0, stream>>>((unsigned*)d_ws);
  neuralode_persistent<<<NWG, TPB, SMEM_BYTES, stream>>>(
      ts, W0, b0, Wh, bh, Wl, bl, betaW, betaB, hvec, scale, y0log,
      (float*)d_out, d_ws);
}
